// Net2_39728447488357
// MI455X (gfx1250) — compile-verified, matrix operand order checked by reading, not run on hardware
//
#include <hip/hip_runtime.h>

// ---------------------------------------------------------------------------
// GCN-ish pipeline for MI455X (gfx1250, wave32).
// - A (8192^2) cast to bf16 once -> 128MB, L2-resident across all five
//   A-multiplies (192MB L2); HBM traffic for A collapses to one read.
// - All matmuls on v_wmma_f32_16x16x32_bf16 with fp32 accumulation.
// - Big GEMMs: 64x64 workgroup tile, 8 waves, 2 accumulators/wave
//   (2 WMMAs per K-chunk per wave between barriers).
// - A tiles moved Global->LDS by the Tensor Data Mover
//   (tensor_load_to_lds + s_wait_tensorcnt), manual fallback otherwise.
// ---------------------------------------------------------------------------

#define NN    8192
#define NFEAT 256
#define NHID1 128
#define NHID2 64
#define KC    32

typedef __attribute__((ext_vector_type(16))) __bf16 v16bf;
typedef __attribute__((ext_vector_type(8)))  float  v8f;
typedef __attribute__((ext_vector_type(4)))  unsigned int u32x4;
typedef __attribute__((ext_vector_type(8)))  int i32x8;
typedef __attribute__((ext_vector_type(4)))  int i32x4;

#if defined(__HIP_DEVICE_COMPILE__) && __has_builtin(__builtin_amdgcn_tensor_load_to_lds) && __has_builtin(__builtin_amdgcn_s_wait_tensorcnt)
#define USE_TDM 1
#else
#define USE_TDM 0
#endif

struct alignas(16) U4 { unsigned int a, b, c, d; };
struct alignas(8)  U2 { unsigned int a, b; };
struct alignas(16) F4 { float a, b, c, d; };

union FragAB { U4 q[2]; v16bf v; };

__device__ __forceinline__ unsigned short f2bf(float f) {
  unsigned u = __builtin_bit_cast(unsigned int, f);
  u += 0x7FFFu + ((u >> 16) & 1u);         // round-to-nearest-even
  return (unsigned short)(u >> 16);
}

#if USE_TDM
// Build a 2D Tensor-DMA descriptor (D#) and issue TENSOR_LOAD_TO_LDS.
// tile_d0 = contiguous K elements, tile_d1 = rows, stride in elements (bf16).
// 6-arg builtin on this toolchain:
//   (u32x4 g0, i32x8 g1, i32x4 g2, i32x4 g3, i32x8 extra, i32 cpol)
__device__ __forceinline__ void tdm_load_2d(unsigned lds_addr,
                                            const void* gaddr,
                                            unsigned tensor_d0, unsigned tensor_d1,
                                            unsigned tile_d0, unsigned tile_d1,
                                            unsigned stride_elems)
{
  unsigned long long ga = (unsigned long long)(size_t)gaddr;
  u32x4 g0;
  g0[0] = 1u;                                             // count=1, user mode
  g0[1] = lds_addr;                                       // lds_addr[63:32]
  g0[2] = (unsigned)(ga & 0xFFFFFFFFu);                   // global_addr lo
  g0[3] = ((unsigned)(ga >> 32) & 0x01FFFFFFu) | (2u << 30); // addr hi | type=2
  i32x8 g1;
  g1[0] = (int)(1u << 16);                                // data_size=1 (2B)
  g1[1] = (int)((tensor_d0 & 0xFFFFu) << 16);             // tensor_dim0 lo16
  g1[2] = (int)(((tensor_d0 >> 16) & 0xFFFFu) | ((tensor_d1 & 0xFFFFu) << 16));
  g1[3] = (int)(((tensor_d1 >> 16) & 0xFFFFu) | ((tile_d0 & 0xFFFFu) << 16));
  g1[4] = (int)(tile_d1 & 0xFFFFu);                       // tile_dim1 (tile_dim2=0)
  g1[5] = (int)stride_elems;                              // tensor_dim0_stride lo32
  g1[6] = 0;
  g1[7] = 0;
  i32x4 z4 = {0, 0, 0, 0};
  i32x8 z8 = {0, 0, 0, 0, 0, 0, 0, 0};
  __builtin_amdgcn_tensor_load_to_lds(g0, g1, z4, z4, z8, 0);
}
#endif

// ---------------------------------------------------------------------------
// Cast fp32 -> bf16, vectorized 4 elements per thread, grid-stride.
// ---------------------------------------------------------------------------
__global__ __launch_bounds__(256) void k_cast(const float* __restrict__ in,
                                              unsigned short* __restrict__ out,
                                              long long n4)
{
  long long i = (long long)blockIdx.x * blockDim.x + threadIdx.x;
  long long stride = (long long)gridDim.x * blockDim.x;
  for (; i < n4; i += stride) {
    F4 f = ((const F4*)in)[i];
    U2 o;
    o.a = (unsigned)f2bf(f.a) | ((unsigned)f2bf(f.b) << 16);
    o.b = (unsigned)f2bf(f.c) | ((unsigned)f2bf(f.d) << 16);
    ((U2*)out)[i] = o;
  }
}

// ---------------------------------------------------------------------------
// Small weight GEMM: Zout[NN,dout] = Xin[NN,din](bf16) @ W[din,dout](fp32),
// output bf16, no activation. Block = 4 waves, each wave owns a 16x16 tile.
// W staged transposed (Wt[n][k]) in LDS once per block.
// ---------------------------------------------------------------------------
__global__ __launch_bounds__(128) void k_wgemm(const unsigned short* __restrict__ Xin,
                                               const float* __restrict__ W,
                                               unsigned short* __restrict__ Zout,
                                               int din, int dout)
{
  __shared__ __align__(16) unsigned short Wt[16 * 256];   // max din = 256

  const int tid  = threadIdx.x;
  const int lane = tid & 31;
  const int wave = tid >> 5;
  const int m0 = blockIdx.x * 64 + wave * 16;
  const int n0 = blockIdx.y * 16;

  for (int e = tid; e < din * 16; e += 128) {
    int k = e >> 4;
    int n = e & 15;
    Wt[n * din + k] = f2bf(W[(size_t)k * dout + n0 + n]);
  }
  __syncthreads();

  v8f acc = {0.f, 0.f, 0.f, 0.f, 0.f, 0.f, 0.f, 0.f};
  const int arow  = m0 + (lane & 15);
  const int ahalf = (lane >> 4) * 16;
  const int bn    = lane & 15;

  for (int k0 = 0; k0 < din; k0 += KC) {
    FragAB fa, fb;
    const unsigned short* ap = Xin + (size_t)arow * din + k0 + ahalf;
    fa.q[0] = *(const U4*)ap;
    fa.q[1] = *(const U4*)(ap + 8);
    const unsigned short* bp = &Wt[bn * din + k0 + ahalf];
    fb.q[0] = *(const U4*)bp;
    fb.q[1] = *(const U4*)(bp + 8);
    acc = __builtin_amdgcn_wmma_f32_16x16x32_bf16(false, fa.v, false, fb.v,
                                                  (short)0, acc, false, false);
  }

  const int colg  = n0 + (lane & 15);
  const int rbase = m0 + ((lane >> 4) << 3);
  #pragma unroll
  for (int r = 0; r < 8; ++r)
    Zout[(size_t)(rbase + r) * dout + colg] = f2bf(acc[r]);
}

// ---------------------------------------------------------------------------
// Big A-multiply: C = relu(A_bf[NN,NN] @ Z[NN,d]).
// Workgroup = 8 waves computing a 64x64 tile: waves form a 4(M) x 2(N) grid,
// each wave owns two 16x16 accumulators (n offsets +0 / +16 within its 32-col
// strip) -> 2 WMMAs per K-chunk per wave between barriers.
// A tile (64xKC) via TDM; B tile (KCx64) staged transposed in LDS.
// ---------------------------------------------------------------------------
__global__ __launch_bounds__(256) void k_agemm(const unsigned short* __restrict__ Abf,
                                               const unsigned short* __restrict__ Z,
                                               float* __restrict__ outF,
                                               unsigned short* __restrict__ outB,
                                               int d, int out_f32)
{
  __shared__ __align__(16) unsigned short sA[64 * KC];    // 4KB
  __shared__ __align__(16) unsigned short sBt[64 * KC];   // 4KB, [n][k]

  const int tid  = threadIdx.x;
  const int lane = tid & 31;
  const int wave = tid >> 5;
  const int wm = wave & 3;                  // 0..3 (M)
  const int wn = wave >> 2;                 // 0..1 (N, 32-col strip)
  const int m0 = blockIdx.x * 64;
  const int n0 = blockIdx.y * 64;

  v8f acc0 = {0.f, 0.f, 0.f, 0.f, 0.f, 0.f, 0.f, 0.f};
  v8f acc1 = {0.f, 0.f, 0.f, 0.f, 0.f, 0.f, 0.f, 0.f};

  const int arow  = wm * 16 + (lane & 15);
  const int ahalf = (lane >> 4) * 16;
  const int bcol0 = wn * 32 + (lane & 15);  // first 16x16 n-subtile
  const int bcol1 = bcol0 + 16;             // second

  for (int k0 = 0; k0 < NN; k0 += KC) {
    __syncthreads();                         // prior tile fully consumed
#if USE_TDM
    if (wave == 0) {
      tdm_load_2d((unsigned)(size_t)sA, Abf + (size_t)m0 * NN + k0,
                  NN, NN, KC, 64, NN);
    }
#else
    {
      int r = tid >> 2;                      // 0..63
      int c = (tid & 3) * 8;                 // 0,8,16,24
      *(U4*)&sA[r * KC + c] =
          *(const U4*)(Abf + (size_t)(m0 + r) * NN + k0 + c);
    }
#endif
    // B tile: rows k0..k0+31, cols n0..n0+63, stored transposed sBt[n][k]
    {
      int r = tid >> 3;                      // k 0..31
      int c = (tid & 7) * 8;                 // n 0..56
      U4 v = *(const U4*)(Z + (size_t)(k0 + r) * d + n0 + c);
      const unsigned short* p = (const unsigned short*)&v;
      #pragma unroll
      for (int j = 0; j < 8; ++j)
        sBt[(c + j) * KC + r] = p[j];
    }
#if USE_TDM
    if (wave == 0) __builtin_amdgcn_s_wait_tensorcnt(0);
#endif
    __syncthreads();

    FragAB fa, fb0, fb1;
    fa.q[0]  = *(const U4*)&sA[arow * KC + ahalf];
    fa.q[1]  = *(const U4*)&sA[arow * KC + ahalf + 8];
    fb0.q[0] = *(const U4*)&sBt[bcol0 * KC + ahalf];
    fb0.q[1] = *(const U4*)&sBt[bcol0 * KC + ahalf + 8];
    fb1.q[0] = *(const U4*)&sBt[bcol1 * KC + ahalf];
    fb1.q[1] = *(const U4*)&sBt[bcol1 * KC + ahalf + 8];
    acc0 = __builtin_amdgcn_wmma_f32_16x16x32_bf16(false, fa.v, false, fb0.v,
                                                   (short)0, acc0, false, false);
    acc1 = __builtin_amdgcn_wmma_f32_16x16x32_bf16(false, fa.v, false, fb1.v,
                                                   (short)0, acc1, false, false);
  }

  const int colg  = n0 + wn * 32 + (lane & 15);
  const int rbase = m0 + wm * 16 + ((lane >> 4) << 3);
  if (out_f32) {
    #pragma unroll
    for (int r = 0; r < 8; ++r) {
      float v0 = acc0[r], v1 = acc1[r];
      outF[(size_t)(rbase + r) * d + colg]      = v0 > 0.f ? v0 : 0.f;
      outF[(size_t)(rbase + r) * d + colg + 16] = v1 > 0.f ? v1 : 0.f;
    }
  } else {
    #pragma unroll
    for (int r = 0; r < 8; ++r) {
      float v0 = acc0[r], v1 = acc1[r];
      outB[(size_t)(rbase + r) * d + colg]      = f2bf(v0 > 0.f ? v0 : 0.f);
      outB[(size_t)(rbase + r) * d + colg + 16] = f2bf(v1 > 0.f ? v1 : 0.f);
    }
  }
}

// ---------------------------------------------------------------------------
// A_rec = sigmoid(S @ S^T), S[NN,NHID1] bf16, 64x64 workgroup tile.
// Both operands are row-tiles of S -> both staged by TDM (waves 0/1).
// sB is [n][k] row-major = rows n0..n0+63 of S, which IS the transposed
// B operand layout for S@S^T.
// ---------------------------------------------------------------------------
__global__ __launch_bounds__(256) void k_sst(const unsigned short* __restrict__ S,
                                             float* __restrict__ out)
{
  __shared__ __align__(16) unsigned short sA[64 * KC];
  __shared__ __align__(16) unsigned short sB[64 * KC];

  const int tid  = threadIdx.x;
  const int lane = tid & 31;
  const int wave = tid >> 5;
  const int wm = wave & 3;
  const int wn = wave >> 2;
  const int m0 = blockIdx.x * 64;
  const int n0 = blockIdx.y * 64;

  v8f acc0 = {0.f, 0.f, 0.f, 0.f, 0.f, 0.f, 0.f, 0.f};
  v8f acc1 = {0.f, 0.f, 0.f, 0.f, 0.f, 0.f, 0.f, 0.f};

  const int arow  = wm * 16 + (lane & 15);
  const int ahalf = (lane >> 4) * 16;
  const int bcol0 = wn * 32 + (lane & 15);
  const int bcol1 = bcol0 + 16;

  for (int k0 = 0; k0 < NHID1; k0 += KC) {
    __syncthreads();
#if USE_TDM
    if (wave == 0) {
      tdm_load_2d((unsigned)(size_t)sA, S + (size_t)m0 * NHID1 + k0,
                  NHID1, NN, KC, 64, NHID1);
    } else if (wave == 1) {
      tdm_load_2d((unsigned)(size_t)sB, S + (size_t)n0 * NHID1 + k0,
                  NHID1, NN, KC, 64, NHID1);
    }
    if (wave < 2) __builtin_amdgcn_s_wait_tensorcnt(0);
#else
    {
      int r = tid >> 2;                      // 0..63
      int c = (tid & 3) * 8;
      *(U4*)&sA[r * KC + c] =
          *(const U4*)(S + (size_t)(m0 + r) * NHID1 + k0 + c);
      *(U4*)&sB[r * KC + c] =
          *(const U4*)(S + (size_t)(n0 + r) * NHID1 + k0 + c);
    }
#endif
    __syncthreads();

    FragAB fa, fb0, fb1;
    fa.q[0]  = *(const U4*)&sA[arow * KC + ahalf];
    fa.q[1]  = *(const U4*)&sA[arow * KC + ahalf + 8];
    fb0.q[0] = *(const U4*)&sB[bcol0 * KC + ahalf];
    fb0.q[1] = *(const U4*)&sB[bcol0 * KC + ahalf + 8];
    fb1.q[0] = *(const U4*)&sB[bcol1 * KC + ahalf];
    fb1.q[1] = *(const U4*)&sB[bcol1 * KC + ahalf + 8];
    acc0 = __builtin_amdgcn_wmma_f32_16x16x32_bf16(false, fa.v, false, fb0.v,
                                                   (short)0, acc0, false, false);
    acc1 = __builtin_amdgcn_wmma_f32_16x16x32_bf16(false, fa.v, false, fb1.v,
                                                   (short)0, acc1, false, false);
  }

  const int colg  = n0 + wn * 32 + (lane & 15);
  const int rbase = m0 + wm * 16 + ((lane >> 4) << 3);
  #pragma unroll
  for (int r = 0; r < 8; ++r) {
    float v0 = acc0[r], v1 = acc1[r];
    out[(size_t)(rbase + r) * NN + colg]      = 1.f / (1.f + __expf(-v0));
    out[(size_t)(rbase + r) * NN + colg + 16] = 1.f / (1.f + __expf(-v1));
  }
}

// ---------------------------------------------------------------------------
// Host-side launch sequence.
// ---------------------------------------------------------------------------
extern "C" void kernel_launch(void* const* d_in, const int* in_sizes, int n_in,
                              void* d_out, int out_size, void* d_ws, size_t ws_size,
                              hipStream_t stream)
{
  (void)in_sizes; (void)n_in; (void)out_size; (void)ws_size;

  const float* X      = (const float*)d_in[0];   // [NN, NFEAT]
  const float* A      = (const float*)d_in[1];   // [NN, NN]
  const float* W_enc1 = (const float*)d_in[2];   // [256,128]
  const float* W_enc2 = (const float*)d_in[3];   // [128,64]
  const float* W_att1 = (const float*)d_in[4];   // [64,128]
  const float* W_att2 = (const float*)d_in[5];   // [128,256]
  const float* W_str1 = (const float*)d_in[6];   // [64,128]

  float* out_att  = (float*)d_out;                         // NN*NFEAT
  float* out_arec = out_att + (size_t)NN * NFEAT;          // NN*NN

  // Workspace: A_bf (128MB) + activation buffers + S.
  unsigned short* A_bf = (unsigned short*)d_ws;
  unsigned short* buf0 = A_bf + (size_t)NN * NN;           // X_bf / Att1
  unsigned short* buf1 = buf0 + (size_t)NN * 256;          // Z ping
  unsigned short* buf2 = buf1 + (size_t)NN * 256;          // H pong
  unsigned short* S_bf = buf2 + (size_t)NN * 256;          // [NN, NHID1]

  // 1) casts: A (L2-resident bf16 copy), X
  k_cast<<<8192, 256, 0, stream>>>(A, A_bf, (long long)NN * NN / 4);
  k_cast<<<2048, 256, 0, stream>>>(X, buf0, (long long)NN * NFEAT / 4);

  // 2) Z1 = X @ W_enc1 ; H1 = relu(A @ Z1)
  k_wgemm<<<dim3(NN/64, NHID1/16), 128, 0, stream>>>(buf0, W_enc1, buf1, NFEAT, NHID1);
  k_agemm<<<dim3(NN/64, NHID1/64), 256, 0, stream>>>(A_bf, buf1, nullptr, buf2, NHID1, 0);

  // 3) Z2 = H1 @ W_enc2 ; H2 = relu(A @ Z2)
  k_wgemm<<<dim3(NN/64, NHID2/16), 128, 0, stream>>>(buf2, W_enc2, buf1, NHID1, NHID2);
  k_agemm<<<dim3(NN/64, NHID2/64), 256, 0, stream>>>(A_bf, buf1, nullptr, buf2, NHID2, 0);

  // 4) S = relu(A @ (H2 @ W_str1))
  k_wgemm<<<dim3(NN/64, NHID1/16), 128, 0, stream>>>(buf2, W_str1, buf1, NHID2, NHID1);
  k_agemm<<<dim3(NN/64, NHID1/64), 256, 0, stream>>>(A_bf, buf1, nullptr, S_bf, NHID1, 0);

  // 5) Att1 = relu(A @ (H2 @ W_att1))
  k_wgemm<<<dim3(NN/64, NHID1/16), 128, 0, stream>>>(buf2, W_att1, buf1, NHID2, NHID1);
  k_agemm<<<dim3(NN/64, NHID1/64), 256, 0, stream>>>(A_bf, buf1, nullptr, buf0, NHID1, 0);

  // 6) Att = relu(A @ (Att1 @ W_att2)) -> fp32 output
  k_wgemm<<<dim3(NN/64, NFEAT/16), 128, 0, stream>>>(buf0, W_att2, buf1, NHID1, NFEAT);
  k_agemm<<<dim3(NN/64, NFEAT/64), 256, 0, stream>>>(A_bf, buf1, out_att, nullptr, NFEAT, 1);

  // 7) A_rec = sigmoid(S @ S^T) -> fp32 output
  k_sst<<<dim3(NN/64, NN/64), 256, 0, stream>>>(S_bf, out_arec);
}
